// CrossAttention_3994319585492
// MI455X (gfx1250) — matrix operand order repaired and checked by reading, not compiled, not run on hardware
//
#include <hip/hip_runtime.h>
#include <hip/hip_bf16.h>
#include <math.h>

typedef __attribute__((ext_vector_type(16))) _Float16 v16h;
typedef __attribute__((ext_vector_type(8)))  _Float16 v8h;
typedef __attribute__((ext_vector_type(4)))  _Float16 v4h;
typedef __attribute__((ext_vector_type(8)))  float    v8f;

#define EMBED 1024
#define CROSS 768
#define NHEAD 16
#define HDIM  64
#define BATCH 8
#define SX    4096
#define SY    77
#define SYP   96          // padded key length: 3 WMMA K-steps of 32
#define MROWS (BATCH*SX)  // 32768

static __device__ inline v8f wmma_f16(v16h a, v16h b, v8f c) {
  // v_wmma_f32_16x16x32_f16: D = A(16x32 f16) * B(32x16 f16) + C(16x16 f32)
  return __builtin_amdgcn_wmma_f32_16x16x32_f16(false, a, false, b, (short)0, c,
                                                false, false);
}

// A-fragment (16x32 f16) per ISA 7.12.2:
// lane L: m = L&15, half = L>>4
//   elems 0..7  (VGPR0..3): K = k0 + half*8 + 0..7
//   elems 8..15 (VGPR4..7): K = k0 + 16 + half*8 + 0..7
static __device__ inline v16h ld_afrag(const _Float16* base, int stride, int m,
                                       int k0, int half) {
  union { v16h v; v8h h[2]; } u;
  const _Float16* p = base + m * stride + k0 + half * 8;
  u.h[0] = *(const v8h*)(p);
  u.h[1] = *(const v8h*)(p + 16);
  return u.v;
}

// B-fragment (32x16 f16): lane L: n = L&15, half = L>>4
//   elems 0..7: K = k0 + half*8 + 0..7; elems 8..15: K = k0 + 16 + half*8 + 0..7   (read from [n][k]-transposed tile)
static __device__ inline v16h ld_bfrag(const _Float16* baseT, int stride, int n,
                                       int k0, int half) {
  union { v16h v; v8h h[2]; } u;
  const _Float16* p = baseT + n * stride + k0 + half * 8;
  u.h[0] = *(const v8h*)(p);
  u.h[1] = *(const v8h*)(p + 16);
  return u.v;
}

// ---------------------------------------------------------------------------
// Tiled WMMA GEMM, double-buffered LDS: Out[M,N] = A[M,K]*W[K,N] + bias.
// Block 128(M) x 128(N), BK=32. 8 waves, each computes a 16x128 strip
// (8 WMMAs per A-fragment). Requires M % 128 == 0, N % 128 == 0, K % 32 == 0.
// Pipeline per K-tile: issue global loads for tile k+1 -> WMMA chain on tile k
// -> convert+ds_store tile k+1 -> one barrier.
// ---------------------------------------------------------------------------
template <bool A_F16, bool OUT_F16>
__global__ __launch_bounds__(256) void gemm_wmma(const void* __restrict__ Ain,
                                                 const float* __restrict__ W,
                                                 const float* __restrict__ bias,
                                                 void* __restrict__ Out, int M,
                                                 int N, int K) {
  constexpr int BM = 128, BN = 128, BK = 32, LDA = BK + 8, LDB = BK + 8;
  constexpr int ASZ = BM * LDA, BSZ = BN * LDB;
  __shared__ __align__(16) _Float16 As[2 * ASZ];  // [m][k]
  __shared__ __align__(16) _Float16 Bs[2 * BSZ];  // [n][k] (transposed)
  const int tid = threadIdx.x, wave = tid >> 5, lane = tid & 31;
  const int half = lane >> 4, lm = lane & 15;
  const int bm = blockIdx.x, bn = blockIdx.y;
  v8f acc[8] = {};
  const int ktiles = K / BK;

  float4 fa[4];  // fp32 A staging regs
  v8h   ha[2];   // f16  A staging regs
  float4 fw[4];  // W staging regs

  auto loadA = [&](int k0g) {
    if (A_F16) {
      const _Float16* A = (const _Float16*)Ain;
#pragma unroll
      for (int i = 0; i < 2; i++) {  // 128*32 f16 / 8 per chunk / 256 thr
        int c = tid + i * 256;
        int row = c >> 2, coff = (c & 3) * 8;
        ha[i] = *(const v8h*)(A + (size_t)(bm * BM + row) * K + k0g + coff);
      }
    } else {
      const float* A = (const float*)Ain;
#pragma unroll
      for (int i = 0; i < 4; i++) {  // 128*32 f32 / 4 per chunk / 256 thr
        int c = tid + i * 256;
        int row = c >> 3, coff = (c & 7) * 4;
        fa[i] = *(const float4*)(A + (size_t)(bm * BM + row) * K + k0g + coff);
      }
    }
  };
  auto loadW = [&](int k0g) {
#pragma unroll
    for (int i = 0; i < 4; i++) {  // 32*128 f32 / 4 per chunk / 256 thr
      int c = tid + i * 256;
      int kr = c >> 5, noff = (c & 31) * 4;
      fw[i] = *(const float4*)(W + (size_t)(k0g + kr) * N + bn * BN + noff);
    }
  };
  auto storeA = [&](int buf) {
    _Float16* dst = As + buf * ASZ;
    if (A_F16) {
#pragma unroll
      for (int i = 0; i < 2; i++) {
        int c = tid + i * 256;
        int row = c >> 2, coff = (c & 3) * 8;
        *(v8h*)(dst + row * LDA + coff) = ha[i];
      }
    } else {
#pragma unroll
      for (int i = 0; i < 4; i++) {
        int c = tid + i * 256;
        int row = c >> 3, coff = (c & 7) * 4;
        v4h h = {(_Float16)fa[i].x, (_Float16)fa[i].y, (_Float16)fa[i].z,
                 (_Float16)fa[i].w};
        *(v4h*)(dst + row * LDA + coff) = h;
      }
    }
  };
  auto storeW = [&](int buf) {
    _Float16* dst = Bs + buf * BSZ;
#pragma unroll
    for (int i = 0; i < 4; i++) {
      int c = tid + i * 256;
      int kr = c >> 5, noff = (c & 31) * 4;
      dst[(noff + 0) * LDB + kr] = (_Float16)fw[i].x;
      dst[(noff + 1) * LDB + kr] = (_Float16)fw[i].y;
      dst[(noff + 2) * LDB + kr] = (_Float16)fw[i].z;
      dst[(noff + 3) * LDB + kr] = (_Float16)fw[i].w;
    }
  };

  // Prologue: stage tile 0.
  loadA(0);
  loadW(0);
  storeA(0);
  storeW(0);
  __syncthreads();

  for (int kt = 0; kt < ktiles; ++kt) {
    const int buf = kt & 1;
    // Issue next tile's global loads before the WMMA chain (overlap HBM).
    if (kt + 1 < ktiles) {
      loadA((kt + 1) * BK);
      loadW((kt + 1) * BK);
    }
    if (kt + 2 < ktiles)  // -> global_prefetch_b8
      __builtin_prefetch(
          W + (size_t)((kt + 2) * BK + (tid >> 3)) * N + bn * BN +
              (tid & 7) * 16,
          0, 1);
    const _Float16* Ab = As + buf * ASZ;
    const _Float16* Bb = Bs + buf * BSZ;
    v16h a = ld_afrag(Ab, LDA, wave * 16 + lm, 0, half);
#pragma unroll
    for (int nt = 0; nt < 8; ++nt) {
      v16h b = ld_bfrag(Bb, LDB, nt * 16 + lm, 0, half);
      acc[nt] = wmma_f16(a, b, acc[nt]);
    }
    if (kt + 1 < ktiles) {
      storeA(buf ^ 1);
      storeW(buf ^ 1);
    }
    __syncthreads();
  }

  // Epilogue: C/D layout — VGPR r: lanes 0-15 -> M=r, lanes 16-31 -> M=r+8
#pragma unroll
  for (int nt = 0; nt < 8; ++nt) {
    int n = bn * BN + nt * 16 + lm;
    float bval = bias[n];
#pragma unroll
    for (int r = 0; r < 8; r++) {
      int row = bm * BM + wave * 16 + (half ? r + 8 : r);
      float v = acc[nt][r] + bval;
      if (OUT_F16)
        ((_Float16*)Out)[(size_t)row * N + n] = (_Float16)v;
      else
        ((float*)Out)[(size_t)row * N + n] = v;
    }
  }
}

// ---------------------------------------------------------------------------
// K/V projection: rows are (b, s) with s padded 77 -> 96 (zeros via select,
// loads kept unconditional on clamped addresses -> no exec divergence).
// z==0: K -> [B, 96, 1024] row-major (serves as B^T tile for scores)
// z==1: V -> [B, H, 64, 96] transposed (serves as B^T tile for ctx)
// ---------------------------------------------------------------------------
__global__ __launch_bounds__(256) void proj_kv(
    const float* __restrict__ y, const float* __restrict__ Wk,
    const float* __restrict__ bk, const float* __restrict__ Wv,
    const float* __restrict__ bv, _Float16* __restrict__ Kout,
    _Float16* __restrict__ Vtout) {
  constexpr int BM = 128, BN = 64, BK = 64, LDA = BK + 8, LDB = BK + 8;
  __shared__ __align__(16) _Float16 As[BM * LDA];
  __shared__ __align__(16) _Float16 Bs[BN * LDB];
  const int tid = threadIdx.x, wave = tid >> 5, lane = tid & 31;
  const int half = lane >> 4, lm = lane & 15;
  const int bm = blockIdx.x, bn = blockIdx.y, z = blockIdx.z;
  const float* W = z ? Wv : Wk;
  const float* bias = z ? bv : bk;
  v8f acc[4] = {};
  const int K = CROSS, N = EMBED;
  float4 fa[8], fw[4];
  bool va[8];
  for (int kt = 0; kt < K / BK; ++kt) {
    const int k0g = kt * BK;
    // Load phase (unconditional, clamped row).
#pragma unroll
    for (int i = 0; i < 8; i++) {
      int c = tid + i * 256;
      int row = c >> 4, coff = (c & 15) * 4;
      int rp = bm * BM + row;
      int bb = rp / SYP, s = rp % SYP;
      int scl = s < SY ? s : (SY - 1);
      va[i] = (s < SY);
      fa[i] = *(const float4*)(y + ((size_t)bb * SY + scl) * CROSS + k0g + coff);
    }
#pragma unroll
    for (int i = 0; i < 4; i++) {
      int c = tid + i * 256;
      int kr = c >> 4, noff = (c & 15) * 4;
      fw[i] = *(const float4*)(W + (size_t)(k0g + kr) * N + bn * BN + noff);
    }
    // Store phase.
#pragma unroll
    for (int i = 0; i < 8; i++) {
      int c = tid + i * 256;
      int row = c >> 4, coff = (c & 15) * 4;
      v4h h = {};
      if (va[i])
        h = (v4h){(_Float16)fa[i].x, (_Float16)fa[i].y, (_Float16)fa[i].z,
                  (_Float16)fa[i].w};
      *(v4h*)(&As[row * LDA + coff]) = h;
    }
#pragma unroll
    for (int i = 0; i < 4; i++) {
      int c = tid + i * 256;
      int kr = c >> 4, noff = (c & 15) * 4;
      Bs[(noff + 0) * LDB + kr] = (_Float16)fw[i].x;
      Bs[(noff + 1) * LDB + kr] = (_Float16)fw[i].y;
      Bs[(noff + 2) * LDB + kr] = (_Float16)fw[i].z;
      Bs[(noff + 3) * LDB + kr] = (_Float16)fw[i].w;
    }
    __syncthreads();
    const int m0 = wave * 16;
#pragma unroll
    for (int kk = 0; kk < BK; kk += 32) {
      v16h a = ld_afrag(As, LDA, m0 + lm, kk, half);
#pragma unroll
      for (int nt = 0; nt < 4; ++nt) {
        v16h b = ld_bfrag(Bs, LDB, nt * 16 + lm, kk, half);
        acc[nt] = wmma_f16(a, b, acc[nt]);
      }
    }
    __syncthreads();
  }
#pragma unroll
  for (int nt = 0; nt < 4; ++nt) {
    int n = bn * BN + nt * 16 + lm;
    float bval = bias[n];
#pragma unroll
    for (int r = 0; r < 8; r++) {
      int rp = bm * BM + wave * 16 + (half ? r + 8 : r);
      int bb = rp / SYP, s = rp % SYP;
      float v = acc[nt][r] + bval;
      if (z == 0) {
        Kout[((size_t)bb * SYP + s) * EMBED + n] = (_Float16)v;
      } else {
        int hh = n >> 6, d = n & 63;
        Vtout[(((size_t)bb * NHEAD + hh) * HDIM + d) * SYP + s] = (_Float16)v;
      }
    }
  }
}

// ---------------------------------------------------------------------------
// Attention: grid.x = B*H (128), grid.y = Sx/128 (32). 8 waves x 16 q-rows.
// scores (5 key tiles x 2 K-steps = 10 WMMA), softmax (shfl_xor over 16-lane
// group, wave32), P -> LDS (f16, A-fragment layout), ctx (3 K-steps x 4 d
// tiles = 12 WMMA).
// ---------------------------------------------------------------------------
__global__ __launch_bounds__(256) void attn_kernel(
    const _Float16* __restrict__ Q, const _Float16* __restrict__ Kb,
    const _Float16* __restrict__ Vt, _Float16* __restrict__ Ctx) {
  __shared__ __align__(16) _Float16 Ks[SYP * 72];      // [s][d]
  __shared__ __align__(16) _Float16 Vs[HDIM * 104];    // [d][s]
  __shared__ __align__(16) _Float16 Ps[8 * 16 * 104];  // per-wave P strips
  const int tid = threadIdx.x, wave = tid >> 5, lane = tid & 31;
  const int half = lane >> 4, lm = lane & 15;
  const int bh = blockIdx.x, b = bh >> 4, h = bh & 15, qb = blockIdx.y;

  // Stage K[s][d] (96 x 64 f16)
#pragma unroll
  for (int i = 0; i < 3; i++) {
    int c = tid + i * 256;
    int s = c >> 3, off = (c & 7) * 8;
    *(v8h*)(&Ks[s * 72 + off]) =
        *(const v8h*)(Kb + ((size_t)b * SYP + s) * EMBED + h * HDIM + off);
  }
  // Stage V^T[d][s] (64 x 96 f16)
#pragma unroll
  for (int i = 0; i < 3; i++) {
    int c = tid + i * 256;
    int d = c / 12, off = (c % 12) * 8;
    *(v8h*)(&Vs[d * 104 + off]) =
        *(const v8h*)(Vt + ((size_t)bh * HDIM + d) * SYP + off);
  }
  __syncthreads();

  const size_t qrow0 = (size_t)b * SX + (size_t)qb * 128 + wave * 16;
  // Q fragments straight from global (f16, same k order as ld_afrag)
  v16h aq[2];
#pragma unroll
  for (int kk = 0; kk < 2; ++kk) {
    union { v16h v; v8h hh[2]; } u;
    const _Float16* p =
        Q + (qrow0 + lm) * EMBED + h * HDIM + kk * 32 + half * 8;
    u.hh[0] = *(const v8h*)p;
    u.hh[1] = *(const v8h*)(p + 16);
    aq[kk] = u.v;
  }

  // scores: S = Q * K^T   (B operand [d][s] is Ks read as [n=s][k=d])
  v8f sc[5] = {};
#pragma unroll
  for (int nt = 0; nt < 5; ++nt)
#pragma unroll
    for (int kk = 0; kk < 2; ++kk) {
      v16h bf = ld_bfrag(Ks, 72, nt * 16 + lm, kk * 32, half);
      sc[nt] = wmma_f16(aq[kk], bf, sc[nt]);
    }

  const float scale = 0.125f;  // 1/sqrt(64)
#pragma unroll
  for (int r = 0; r < 8; r++) {
    float mx = -3.0e30f;
#pragma unroll
    for (int nt = 0; nt < 5; ++nt) {
      float v = sc[nt][r] * scale;
      if (nt * 16 + lm >= SY) v = -3.0e30f;  // mask padded keys
      sc[nt][r] = v;
      mx = fmaxf(mx, v);
    }
    mx = fmaxf(mx, __shfl_xor(mx, 8, 32));
    mx = fmaxf(mx, __shfl_xor(mx, 4, 32));
    mx = fmaxf(mx, __shfl_xor(mx, 2, 32));
    mx = fmaxf(mx, __shfl_xor(mx, 1, 32));
    float sum = 0.f;
#pragma unroll
    for (int nt = 0; nt < 5; ++nt) {
      float p = __expf(sc[nt][r] - mx);
      sc[nt][r] = p;
      sum += p;
    }
    sum += __shfl_xor(sum, 8, 32);
    sum += __shfl_xor(sum, 4, 32);
    sum += __shfl_xor(sum, 2, 32);
    sum += __shfl_xor(sum, 1, 32);
    float inv = 1.0f / sum;
    int prow = wave * 16 + (half ? r + 8 : r);
#pragma unroll
    for (int nt = 0; nt < 5; ++nt)
      Ps[prow * 104 + nt * 16 + lm] = (_Float16)(sc[nt][r] * inv);
    Ps[prow * 104 + 80 + lm] = (_Float16)0.f;  // zero pad cols 80..95
  }

  // ctx = P * V  (A from per-wave Ps strip, same-wave LDS RAW -> hw DScnt)
  v8f ac[4] = {};
#pragma unroll
  for (int kk = 0; kk < 3; ++kk) {
    v16h a = ld_afrag(Ps, 104, wave * 16 + lm, kk * 32, half);
#pragma unroll
    for (int nt = 0; nt < 4; ++nt) {
      v16h bf = ld_bfrag(Vs, 104, nt * 16 + lm, kk * 32, half);
      ac[nt] = wmma_f16(a, bf, ac[nt]);
    }
  }
#pragma unroll
  for (int nt = 0; nt < 4; ++nt) {
    int col = h * HDIM + nt * 16 + lm;
#pragma unroll
    for (int r = 0; r < 8; r++) {
      size_t row = (size_t)b * SX + (size_t)qb * 128 + wave * 16 +
                   (half ? r + 8 : r);
      Ctx[row * EMBED + col] = (_Float16)ac[nt][r];
    }
  }
}

// ---------------------------------------------------------------------------
extern "C" void kernel_launch(void* const* d_in, const int* in_sizes, int n_in,
                              void* d_out, int out_size, void* d_ws,
                              size_t ws_size, hipStream_t stream) {
  const float* x  = (const float*)d_in[0];
  const float* y  = (const float*)d_in[1];
  const float* Wq = (const float*)d_in[2];
  const float* bq = (const float*)d_in[3];
  const float* Wk = (const float*)d_in[4];
  const float* bk = (const float*)d_in[5];
  const float* Wv = (const float*)d_in[6];
  const float* bv = (const float*)d_in[7];
  const float* Wo = (const float*)d_in[8];
  const float* bo = (const float*)d_in[9];
  float* out = (float*)d_out;

  // Workspace layout (f16 intermediates):
  //   Q   [32768,1024]        = 64 MiB
  //   K   [8,96,1024]         = 1.5 MiB
  //   V^T [8,16,64,96]        = 1.5 MiB
  //   ctx [32768,1024]        = 64 MiB
  char* ws = (char*)d_ws;
  const size_t QB = (size_t)MROWS * EMBED * 2;        // 67108864
  const size_t KB = (size_t)BATCH * SYP * EMBED * 2;  // 1572864
  _Float16* Qbuf  = (_Float16*)(ws);
  _Float16* Kbuf  = (_Float16*)(ws + QB);
  _Float16* Vtbuf = (_Float16*)(ws + QB + KB);
  _Float16* Ctxb  = (_Float16*)(ws + QB + 2 * KB);

  dim3 blk(256);
  // 1) Q = x @ Wq + bq   -> f16
  gemm_wmma<false, true><<<dim3(MROWS / 128, EMBED / 128), blk, 0, stream>>>(
      x, Wq, bq, Qbuf, MROWS, EMBED, EMBED);
  // 2) K, V projections (padded to 96 rows per batch)
  proj_kv<<<dim3((BATCH * SYP) / 128, EMBED / 64, 2), blk, 0, stream>>>(
      y, Wk, bk, Wv, bv, Kbuf, Vtbuf);
  // 3) attention -> ctx f16
  attn_kernel<<<dim3(BATCH * NHEAD, SX / 128), blk, 0, stream>>>(Qbuf, Kbuf,
                                                                 Vtbuf, Ctxb);
  // 4) out = ctx @ Wo + bo -> f32
  gemm_wmma<true, false><<<dim3(MROWS / 128, EMBED / 128), blk, 0, stream>>>(
      Ctxb, Wo, bo, out, MROWS, EMBED, EMBED);
}
